// OptimizedDualAttention_38122129719510
// MI455X (gfx1250) — matrix operand order repaired and checked by reading, not compiled, not run on hardware
//
#include <hip/hip_runtime.h>
#include <math.h>

// ---------------------------------------------------------------------------
// OptimizedDualAttention for MI455X (gfx1250): WMMA f16 GEMMs + fused
// top-32 attention with LDS-resident score strips.
//   B=4, S=2048, IN=256, E=128, H=4, d=32, TOPK=32
// ---------------------------------------------------------------------------

typedef __attribute__((ext_vector_type(16))) _Float16 v16h;
typedef __attribute__((ext_vector_type(8)))  _Float16 v8h;
typedef __attribute__((ext_vector_type(8)))  float    v8f;

#define BATCH   4
#define SEQ     2048
#define INDIM   256
#define EDIM    128
#define HEADS   4
#define HDIM    32
#define TOPK    32
#define MROWS   (BATCH*SEQ)        // 8192

// ----- WMMA fragment loaders (CDNA5 16x16x32 f16 layouts, wave32) -----------

// A fragment: 16x32 tile of row-major A (lda in halves), tile origin passed in.
// lane<16 : row M=lane,    halves = K[0..7],  K[16..23]
// lane>=16: row M=lane-16, halves = K[8..15], K[24..31]
__device__ inline v16h load_a16x32(const _Float16* __restrict__ A, int lda, int lane) {
  const int m   = lane & 15;
  const int sel = lane >> 4;           // 0 or 1
  const _Float16* p = A + m * lda + sel * 8;
  v8h lo = *(const v8h*)(p);
  v8h hi = *(const v8h*)(p + 16);
  v16h r;
#pragma unroll
  for (int i = 0; i < 8; ++i) { r[i] = lo[i]; r[i + 8] = hi[i]; }
  return r;
}

// B fragment: 32x16 tile of B where B[k][n] = W[n][k], W row-major [N,K]
// (ldw in halves). Exactly matches torch Linear weight storage -> x @ W.T.
// lane<16 : col N=lane,    halves = K[0..7],  K[16..23]
// lane>=16: col N=lane-16, halves = K[8..15], K[24..31]
__device__ inline v16h load_b32x16(const _Float16* __restrict__ W, int ldw, int lane) {
  const int n    = lane & 15;
  const int koff = (lane >> 4) * 8;
  const _Float16* p = W + n * ldw + koff;
  v8h lo = *(const v8h*)(p);
  v8h hi = *(const v8h*)(p + 16);
  v16h r;
#pragma unroll
  for (int i = 0; i < 8; ++i) { r[i] = lo[i]; r[i + 8] = hi[i]; }
  return r;
}

__device__ inline v8f wmma_f16(v16h a, v16h b, v8f c) {
  return __builtin_amdgcn_wmma_f32_16x16x32_f16(
      false, a, false, b, (short)0, c, false, false);
}

// ----- f32 -> f16 conversion ------------------------------------------------
__global__ void cvt_f32_f16(const float* __restrict__ src,
                            _Float16* __restrict__ dst, int n) {
  int i = blockIdx.x * blockDim.x + threadIdx.x;
  if (i < n) dst[i] = (_Float16)src[i];
}

// ----- WMMA GEMM: C[M,128] = scale * (A[M,K] @ W[128,K]^T + bias) -----------
// 4 waves / block; each wave computes a 16x128 output strip: one A-fragment
// load feeds 8 back-to-back independent WMMAs (8 accumulators), so A global
// traffic is 1/8 of the naive tiling and the matrix pipe sees deep bursts.
// Weights (<=64KB f16) stay resident in WGP$/L2 across the whole grid.
template <int F32OUT>
__global__ __launch_bounds__(128)
void gemm_wt(const _Float16* __restrict__ A, int lda,
             const _Float16* __restrict__ W, int ldw,
             const float* __restrict__ bias,
             void* __restrict__ Cout, int ldc,
             int K, float scale) {
  const int lane = threadIdx.x & 31;
  const int wave = threadIdx.x >> 5;
  const int m0 = (blockIdx.x * 4 + wave) << 4;

  v8f acc[8] = {};
  for (int k0 = 0; k0 < K; k0 += 32) {
    if (k0 + 32 < K)
      __builtin_prefetch(A + (size_t)m0 * lda + k0 + 32, 0, 0);
    const v16h a = load_a16x32(A + (size_t)m0 * lda + k0, lda, lane);
#pragma unroll
    for (int nt = 0; nt < 8; ++nt) {
      v16h b = load_b32x16(W + (size_t)(nt * 16) * ldw + k0, ldw, lane);
      acc[nt] = wmma_f16(a, b, acc[nt]);
    }
  }

  const int rbase = (lane >> 4) * 8;   // C rows: r + 8*(lane>=16)
#pragma unroll
  for (int nt = 0; nt < 8; ++nt) {
    const int col = nt * 16 + (lane & 15);
    const float bv = bias[col];
    if (F32OUT) {
      float* C = (float*)Cout;
#pragma unroll
      for (int r = 0; r < 8; ++r)
        C[(size_t)(m0 + rbase + r) * ldc + col] = (acc[nt][r] + bv) * scale;
    } else {
      _Float16* C = (_Float16*)Cout;
#pragma unroll
      for (int r = 0; r < 8; ++r)
        C[(size_t)(m0 + rbase + r) * ldc + col] =
            (_Float16)((acc[nt][r] + bv) * scale);
    }
  }
}

// ----- fused sparse top-32 attention ---------------------------------------
// Grid: B*H*(S/16) blocks of 128 threads (4 waves).
// Phase 1: scores[16q x S] via v_wmma_f32_16x16x32_f16 into LDS (128 KB,
//          well within the 320 KB WGP LDS; never touches HBM with scores).
// Phase 2: each wave exclusively owns 4 query rows: 32 rounds of wave-wide
//          argmax extraction (matches jax.lax.top_k + softmax order) with
//          online softmax; lane == head channel for the V gather. The
//          masking store -> reload is same-wave LDS traffic, which the ISA
//          keeps in-order (DScnt), so no block barrier is needed per round.
__global__ __launch_bounds__(128)
void attn_topk(const _Float16* __restrict__ Q,   // [B*S, E], pre-scaled by 1/sqrt(d)
               const _Float16* __restrict__ Km,  // [B*S, E]
               const _Float16* __restrict__ V,   // [B*S, E]
               _Float16* __restrict__ O) {       // [B*S, E]
  extern __shared__ float srow[];                // 16 * SEQ floats = 128 KB
  const int lane = threadIdx.x & 31;
  const int wave = threadIdx.x >> 5;
  const int qblocks = SEQ >> 4;
  const int bh = blockIdx.x / qblocks;           // b*HEADS + h
  const int qb = blockIdx.x % qblocks;
  const int b = bh >> 2, h = bh & 3;

  const _Float16* Qb = Q  + ((size_t)(b * SEQ + qb * 16)) * EDIM + h * HDIM;
  const _Float16* Kb = Km + ((size_t)b * SEQ) * EDIM + h * HDIM;
  const _Float16* Vb = V  + ((size_t)b * SEQ) * EDIM + h * HDIM;

  // Q fragment is reused for all key tiles of this wave.
  const v16h qa = load_a16x32(Qb, EDIM, lane);

  // ---- Phase 1: score strip -> LDS (waves split the key dimension) ----
  for (int kt = wave; kt < (SEQ >> 4); kt += 4) {
    v16h kb = load_b32x16(Kb + (size_t)(kt * 16) * EDIM, EDIM, lane);
    v8f c = {};
    c = wmma_f16(qa, kb, c);
    const int key = kt * 16 + (lane & 15);
    const int rb = (lane >> 4) * 8;
#pragma unroll
    for (int r = 0; r < 8; ++r)
      srow[(rb + r) * SEQ + key] = c[r];
  }
  __syncthreads();

  // ---- Phase 2: per-query top-32 extraction + softmax + V gather ----
  for (int m = wave; m < 16; m += 4) {
    float* row = srow + m * SEQ;
    float acc = 0.f, sumexp = 0.f, smax = 0.f;
    for (int it = 0; it < TOPK; ++it) {
      // lane-local argmax over strided subset (64 candidates/lane)
      float best = -INFINITY; int bidx = 0;
      for (int j = lane; j < SEQ; j += 32) {
        float v = row[j];
        if (v > best) { best = v; bidx = j; }
      }
      // wave-wide argmax (ties -> lowest index, like top_k)
#pragma unroll
      for (int off = 16; off > 0; off >>= 1) {
        float ov = __shfl_xor(best, off, 32);
        int   oi = __shfl_xor(bidx, off, 32);
        if (ov > best || (ov == best && oi < bidx)) { best = ov; bidx = oi; }
      }
      if (it == 0) smax = best;
      if (lane == 0) row[bidx] = -INFINITY;   // remove from candidate set
      const float w = __expf(best - smax);
      sumexp += w;
      acc += w * (float)Vb[(size_t)bidx * EDIM + lane];  // lane = channel
    }
    O[((size_t)(b * SEQ + qb * 16 + m)) * EDIM + h * HDIM + lane] =
        (_Float16)(acc / sumexp);
  }
}

// ----- final gating: out = fus * sigmoid(gate) -----------------------------
__global__ void gate_fuse(const float* __restrict__ g,
                          const float* __restrict__ f,
                          float* __restrict__ out, int n) {
  int i = blockIdx.x * blockDim.x + threadIdx.x;
  if (i < n) out[i] = f[i] * (1.0f / (1.0f + __expf(-g[i])));
}

// ---------------------------------------------------------------------------
extern "C" void kernel_launch(void* const* d_in, const int* in_sizes, int n_in,
                              void* d_out, int out_size, void* d_ws, size_t ws_size,
                              hipStream_t stream) {
  (void)in_sizes; (void)n_in; (void)out_size; (void)ws_size;

  char* ws = (char*)d_ws;
  size_t off = 0;
  auto alloc = [&](size_t bytes) -> char* {
    char* p = ws + off;
    off = (off + bytes + 255) & ~(size_t)255;
    return p;
  };

  // f16 scratch tensors
  _Float16* x_h  = (_Float16*)alloc((size_t)MROWS * INDIM * 2);
  // weights (f16), in dict order of the weight matrices
  const int widx[12] = {1, 3, 5, 7, 9, 11, 13, 15, 17, 19, 21, 23};
  const int wn[12]   = {EDIM*INDIM, EDIM*INDIM,
                        EDIM*EDIM, EDIM*EDIM, EDIM*EDIM, EDIM*EDIM,
                        EDIM*EDIM, EDIM*EDIM, EDIM*EDIM, EDIM*EDIM,
                        EDIM*INDIM, EDIM*INDIM};
  _Float16* wh[12];
  for (int i = 0; i < 12; ++i) wh[i] = (_Float16*)alloc((size_t)wn[i] * 2);

  _Float16* tp_h = (_Float16*)alloc((size_t)MROWS * EDIM * 2);
  _Float16* sp_h = (_Float16*)alloc((size_t)MROWS * EDIM * 2);
  _Float16* qt   = (_Float16*)alloc((size_t)MROWS * EDIM * 2);
  _Float16* kt_  = (_Float16*)alloc((size_t)MROWS * EDIM * 2);
  _Float16* vt   = (_Float16*)alloc((size_t)MROWS * EDIM * 2);
  _Float16* qs   = (_Float16*)alloc((size_t)MROWS * EDIM * 2);
  _Float16* ks   = (_Float16*)alloc((size_t)MROWS * EDIM * 2);
  _Float16* vs   = (_Float16*)alloc((size_t)MROWS * EDIM * 2);
  _Float16* at_h = (_Float16*)alloc((size_t)MROWS * EDIM * 2);
  _Float16* as_h = (_Float16*)alloc((size_t)MROWS * EDIM * 2);
  _Float16* cat_h= (_Float16*)alloc((size_t)MROWS * 2 * EDIM * 2);
  float*    gatef= (float*)   alloc((size_t)MROWS * EDIM * 4);
  float*    fusf = (float*)   alloc((size_t)MROWS * EDIM * 4);

  // ---- conversions to f16 ----
  {
    int n = MROWS * INDIM;
    cvt_f32_f16<<<(n + 255) / 256, 256, 0, stream>>>((const float*)d_in[0], x_h, n);
    for (int i = 0; i < 12; ++i)
      cvt_f32_f16<<<(wn[i] + 255) / 256, 256, 0, stream>>>(
          (const float*)d_in[widx[i]], wh[i], wn[i]);
  }

  const float qscale = 0.17677669529663687f;  // 1/sqrt(32)
  const dim3 gblk(128);
  const dim3 ggrid(MROWS / 64);               // 4 waves/block, 16 rows/wave

  auto gemm16 = [&](const _Float16* A, int lda, const _Float16* W, int ldw,
                    const float* bias, _Float16* C, int ldc, int K, float sc) {
    gemm_wt<0><<<ggrid, gblk, 0, stream>>>(A, lda, W, ldw, bias, (void*)C, ldc, K, sc);
  };
  auto gemm32 = [&](const _Float16* A, int lda, const _Float16* W, int ldw,
                    const float* bias, float* C, int ldc, int K) {
    gemm_wt<1><<<ggrid, gblk, 0, stream>>>(A, lda, W, ldw, bias, (void*)C, ldc, K, 1.0f);
  };

  // ---- branch projections: tp = x@tp_w^T+b ; sp = x@sp_w^T+b ----
  gemm16(x_h, INDIM, wh[0], INDIM, (const float*)d_in[2],  tp_h, EDIM, INDIM, 1.0f);
  gemm16(x_h, INDIM, wh[1], INDIM, (const float*)d_in[4],  sp_h, EDIM, INDIM, 1.0f);

  // ---- q/k/v for temporal branch (q pre-scaled by 1/sqrt(d)) ----
  gemm16(tp_h, EDIM, wh[2], EDIM, (const float*)d_in[6],  qt,  EDIM, EDIM, qscale);
  gemm16(tp_h, EDIM, wh[3], EDIM, (const float*)d_in[8],  kt_, EDIM, EDIM, 1.0f);
  gemm16(tp_h, EDIM, wh[4], EDIM, (const float*)d_in[10], vt,  EDIM, EDIM, 1.0f);
  // ---- q/k/v for semantic branch ----
  gemm16(sp_h, EDIM, wh[6], EDIM, (const float*)d_in[14], qs,  EDIM, EDIM, qscale);
  gemm16(sp_h, EDIM, wh[7], EDIM, (const float*)d_in[16], ks,  EDIM, EDIM, 1.0f);
  gemm16(sp_h, EDIM, wh[8], EDIM, (const float*)d_in[18], vs,  EDIM, EDIM, 1.0f);

  // ---- fused top-32 attention (score strip resident in 128 KB LDS) ----
  {
    dim3 grid(BATCH * HEADS * (SEQ / 16));
    size_t lds = (size_t)16 * SEQ * sizeof(float);
    attn_topk<<<grid, dim3(128), lds, stream>>>(qt, kt_, vt, at_h);
    attn_topk<<<grid, dim3(128), lds, stream>>>(qs, ks, vs, as_h);
  }

  // ---- output projections, written straight into the concat buffer ----
  gemm16(at_h, EDIM, wh[5], EDIM, (const float*)d_in[12], cat_h,        2 * EDIM, EDIM, 1.0f);
  gemm16(as_h, EDIM, wh[9], EDIM, (const float*)d_in[20], cat_h + EDIM, 2 * EDIM, EDIM, 1.0f);

  // ---- gate / fuse GEMMs (f32 out), then sigmoid gating ----
  gemm32(cat_h, 2 * EDIM, wh[10], 2 * EDIM, (const float*)d_in[22], gatef, EDIM, 2 * EDIM);
  gemm32(cat_h, 2 * EDIM, wh[11], 2 * EDIM, (const float*)d_in[24], fusf,  EDIM, 2 * EDIM);

  {
    int n = MROWS * EDIM;
    gate_fuse<<<(n + 255) / 256, 256, 0, stream>>>(gatef, fusf, (float*)d_out, n);
  }
}
